// TorchAttentionMechanism_18425409700363
// MI455X (gfx1250) — matrix operand order repaired and checked by reading, not compiled, not run on hardware
//
#include <hip/hip_runtime.h>
#include <hip/hip_bf16.h>

// ---------------- problem constants (fixed by the reference) ----------------
#define BB 8
#define SS 1024
#define EE 1024
#define HH 16
#define DD 64
#define MTOT (BB * SS) /* 8192 */

typedef __bf16 bf16;
typedef __attribute__((ext_vector_type(16))) __bf16 bf16x16;
typedef __attribute__((ext_vector_type(8)))  __bf16 bf16x8;
typedef __attribute__((ext_vector_type(8)))  float  f32x8;
typedef __attribute__((ext_vector_type(4)))  unsigned int u32x4;
typedef __attribute__((ext_vector_type(8)))  int i32x8;
typedef __attribute__((ext_vector_type(4)))  int i32x4;

static __device__ inline f32x8 wmma_bf16(bf16x16 a, bf16x16 b, f32x8 c) {
  // D = A(16x32) * B(32x16) + C, f32 accumulate
  return __builtin_amdgcn_wmma_f32_16x16x32_bf16(false, a, false, b, (short)0, c,
                                                 false, false);
}

// ---------------- Tensor Data Mover: 2D bf16 tile -> LDS --------------------
// Packs a D# (groups 0/1 per CDNA5 ISA 8.3/8.4) for a [tile_d1 x tile_d0]
// element tile of a row-major tensor with row stride `stride0` elements.
// All inputs are wave-uniform (descriptor lives in SGPRs). EXEC is ignored by
// TDM, so call under a wave-uniform branch from a single wave.
static __device__ inline void tdm_load_2d_bf16(unsigned lds_addr, const void* gaddr,
                                               unsigned tile_d0, unsigned tile_d1,
                                               unsigned long long tensor_d0,
                                               unsigned long long tensor_d1,
                                               unsigned long long stride0) {
  const unsigned long long ga = (unsigned long long)(uintptr_t)gaddr;
  u32x4 g0;
  g0[0] = 1u;                                               // count=1 (valid, user mode)
  g0[1] = lds_addr;                                         // lds_addr [63:32]
  g0[2] = (unsigned)(ga & 0xffffffffu);                     // global_addr lo
  g0[3] = (unsigned)((ga >> 32) & 0x1ffffffu) | (2u << 30); // global_addr hi | type=2
  // group1: bit-packed via 64-bit quads
  const unsigned long long q0 =
      (1ull << 16)                                          // data_size = 2 bytes
    | ((tensor_d0 & 0xffffull) << 48);                      // tensor_dim0[15:0]
  const unsigned long long q1 =
      ((tensor_d0 >> 16) & 0xffffull)                       // tensor_dim0[31:16]
    | ((tensor_d1 & 0xffffffffull) << 16)                   // tensor_dim1
    | ((unsigned long long)(tile_d0 & 0xffffu) << 48);      // tile_dim0
  const unsigned long long q2 =
      (unsigned long long)(tile_d1 & 0xffffu)               // tile_dim1 (tile_dim2=0)
    | ((stride0 & 0xffffffffull) << 32);                    // tensor_dim0_stride[31:0]
  const unsigned long long q3 =
      ((stride0 >> 32) & 0xffffull);                        // tensor_dim0_stride[47:32]
  i32x8 g1;
  g1[0] = (int)(q0 & 0xffffffffull); g1[1] = (int)(q0 >> 32);
  g1[2] = (int)(q1 & 0xffffffffull); g1[3] = (int)(q1 >> 32);
  g1[4] = (int)(q2 & 0xffffffffull); g1[5] = (int)(q2 >> 32);
  g1[6] = (int)(q3 & 0xffffffffull); g1[7] = (int)(q3 >> 32);
  const i32x4 z4 = {0, 0, 0, 0};
#if __has_include(<hip/amd_detail/amd_gfx1250_TDM.h>)
  const i32x8 z8 = {0, 0, 0, 0, 0, 0, 0, 0};
  __builtin_amdgcn_tensor_load_to_lds(g0, g1, z4, z4, z8, 0);
#else
  __builtin_amdgcn_tensor_load_to_lds(g0, g1, z4, z4, 0);
#endif
}

// A-fragment (16x32 MxK). Lane: m = lane&15, kb = lane<16 ? 0 : 8;
// elements 0..7 -> K=kb..kb+7, elements 8..15 -> K=kb+16..kb+23.
static __device__ inline bf16x16 frag_a(const bf16* base, int ld, int k0, int lane) {
  const int m  = lane & 15;
  const int kb = (lane < 16) ? 0 : 8;
  const bf16* p = base + (size_t)m * ld + k0 + kb;
  bf16x8 lo = *(const bf16x8*)(p);
  bf16x8 hi = *(const bf16x8*)(p + 16);
  bf16x16 a;
#pragma unroll
  for (int i = 0; i < 8; ++i) { a[i] = lo[i]; a[i + 8] = hi[i]; }
  return a;
}

// B-fragment (32x16 KxN) read from row-major [N, ld] storage (K contiguous).
// Lane: n = lane&15, kb = lane<16 ? 0 : 8; elements 0..7 -> K=kb..kb+7, 8..15 -> K=kb+16..kb+23.
static __device__ inline bf16x16 frag_b(const bf16* base, int ld, int k0, int lane) {
  const int n  = lane & 15;
  const int kb = (lane < 16) ? 0 : 8;
  const bf16* p = base + (size_t)n * ld + k0 + kb;
  bf16x8 lo = *(const bf16x8*)(p);
  bf16x8 hi = *(const bf16x8*)(p + 16);
  bf16x16 b;
#pragma unroll
  for (int i = 0; i < 8; ++i) { b[i] = lo[i]; b[i + 8] = hi[i]; }
  return b;
}

// ---------------- fp32 -> bf16 conversion ----------------
__global__ __launch_bounds__(256) void cvt_f32_to_bf16(const float* __restrict__ src,
                                                       bf16* __restrict__ dst, int n4) {
  int i = blockIdx.x * blockDim.x + threadIdx.x;
  if (i >= n4) return;
  float4 f = ((const float4*)src)[i];
  union { bf16 h[4]; uint2 u; } pk;
  pk.h[0] = (bf16)f.x; pk.h[1] = (bf16)f.y; pk.h[2] = (bf16)f.z; pk.h[3] = (bf16)f.w;
  ((uint2*)dst)[i] = pk.u;
}

// ---------------- bf16 GEMM: C[M,N] = A[M,K] * B[N,K]^T (y = x @ W^T) -------
// TDM double-buffered: wave 0 DMA-stages the next K-chunk's A/B tiles while
// all 8 waves run WMMA on the current buffer.
#define GBM 128
#define GBN 64
#define GBK 32

__global__ __launch_bounds__(256) void gemm_bf16_nt(const bf16* __restrict__ A,
                                                    const bf16* __restrict__ Bw,
                                                    float* __restrict__ C,
                                                    int M, int N, int K) {
  __shared__ bf16 As[2][GBM * GBK]; // 2 x 8 KB
  __shared__ bf16 Bs[2][GBN * GBK]; // 2 x 4 KB

  const int tid  = threadIdx.x;
  const int lane = tid & 31;
  const int wave = tid >> 5;     // 0..7
  const int wy   = wave & 3;     // 32-row strip
  const int wx   = wave >> 2;    // 32-col strip
  const int bm   = blockIdx.x * GBM;
  const int bn   = blockIdx.y * GBN;

  const bf16* Abase = A  + (size_t)bm * K;
  const bf16* Bbase = Bw + (size_t)bn * K;
  const unsigned asb[2] = {(unsigned)(uintptr_t)(void*)&As[0][0],
                           (unsigned)(uintptr_t)(void*)&As[1][0]};
  const unsigned bsb[2] = {(unsigned)(uintptr_t)(void*)&Bs[0][0],
                           (unsigned)(uintptr_t)(void*)&Bs[1][0]};

  f32x8 acc[2][2];
#pragma unroll
  for (int i = 0; i < 2; ++i)
#pragma unroll
    for (int j = 0; j < 2; ++j) acc[i][j] = (f32x8){0,0,0,0,0,0,0,0};

  if (wave == 0) { // prime buffer 0
    tdm_load_2d_bf16(asb[0], Abase, GBK, GBM, (unsigned)K, (unsigned)M, (unsigned)K);
    tdm_load_2d_bf16(bsb[0], Bbase, GBK, GBN, (unsigned)K, (unsigned)N, (unsigned)K);
  }

  int buf = 0;
  for (int k0 = 0; k0 < K; k0 += GBK) {
    if (wave == 0) {
      if (k0 + GBK < K) { // stage next chunk into the other buffer
        tdm_load_2d_bf16(asb[buf ^ 1], Abase + k0 + GBK, GBK, GBM,
                         (unsigned)K, (unsigned)M, (unsigned)K);
        tdm_load_2d_bf16(bsb[buf ^ 1], Bbase + k0 + GBK, GBK, GBN,
                         (unsigned)K, (unsigned)N, (unsigned)K);
        // TDM completes in order: <=2 outstanding => current buffer done
        __builtin_amdgcn_s_wait_tensorcnt(2);
      } else {
        __builtin_amdgcn_s_wait_tensorcnt(0);
      }
    }
    __syncthreads(); // publish current buffer to all waves

    const bf16* as = &As[buf][0];
    const bf16* bs = &Bs[buf][0];
    bf16x16 bfr[2];
#pragma unroll
    for (int nb = 0; nb < 2; ++nb)
      bfr[nb] = frag_b(bs + (wx * 32 + nb * 16) * GBK, GBK, 0, lane);
#pragma unroll
    for (int mb = 0; mb < 2; ++mb) {
      bf16x16 af = frag_a(as + (wy * 32 + mb * 16) * GBK, GBK, 0, lane);
#pragma unroll
      for (int nb = 0; nb < 2; ++nb) acc[mb][nb] = wmma_bf16(af, bfr[nb], acc[mb][nb]);
    }
    __syncthreads(); // all reads done before buffer is DMA-overwritten
    buf ^= 1;
  }

  const int n  = lane & 15;
  const int mo = (lane >= 16) ? 8 : 0;
#pragma unroll
  for (int mb = 0; mb < 2; ++mb)
#pragma unroll
    for (int nb = 0; nb < 2; ++nb) {
      int cm = bm + wy * 32 + mb * 16 + mo;
      int cn = bn + wx * 32 + nb * 16 + n;
#pragma unroll
      for (int r = 0; r < 8; ++r) C[(size_t)(cm + r) * N + cn] = acc[mb][nb][r];
    }
}

// ------------- per-head centering + variance + V transpose ------------------
// one wave per (b,h,s) row of D=64
__global__ __launch_bounds__(256) void center_heads(
    const float* __restrict__ Qf, const float* __restrict__ Kf,
    const float* __restrict__ Vf,
    bf16* __restrict__ Qc, bf16* __restrict__ Kc, bf16* __restrict__ Vt,
    float* __restrict__ qvar, float* __restrict__ kvar) {
  const int wg   = blockIdx.x * 8 + (threadIdx.x >> 5);
  const int lane = threadIdx.x & 31;
  const int s  = wg & (SS - 1);
  const int bh = wg >> 10;
  const int h  = bh & (HH - 1);
  const int b  = bh >> 4;

  const size_t src = ((size_t)b * SS + s) * EE + (size_t)h * DD;
  const size_t dst = (((size_t)bh) * SS + s) * DD;

#pragma unroll
  for (int which = 0; which < 2; ++which) {
    const float* P = which ? Kf : Qf;
    float f0 = P[src + lane], f1 = P[src + lane + 32];
    float sum = f0 + f1;
#pragma unroll
    for (int off = 16; off > 0; off >>= 1) sum += __shfl_xor(sum, off, 32);
    float mean = sum * (1.0f / 64.0f);
    float c0 = f0 - mean, c1 = f1 - mean;
    float vs = c0 * c0 + c1 * c1;
#pragma unroll
    for (int off = 16; off > 0; off >>= 1) vs += __shfl_xor(vs, off, 32);
    bf16* Cc = which ? Kc : Qc;
    Cc[dst + lane]      = (bf16)c0;
    Cc[dst + lane + 32] = (bf16)c1;
    if (lane == 0) {
      float* Vr = which ? kvar : qvar;
      Vr[(size_t)bh * SS + s] = vs * (1.0f / 63.0f); // ddof=1
    }
  }
  { // V: bf16 + transpose to [B,H,D,S]
    float f0 = Vf[src + lane], f1 = Vf[src + lane + 32];
    const size_t base = ((size_t)bh * DD) * SS + s;
    Vt[base + (size_t)lane * SS]        = (bf16)f0;
    Vt[base + (size_t)(lane + 32) * SS] = (bf16)f1;
  }
}

// ---------------- attention: scores in LDS, two WMMA phases -----------------
#define TQ 32
__global__ __launch_bounds__(256) void attention_kernel(
    const bf16* __restrict__ Qc,   // [B,H,S,D]
    const bf16* __restrict__ Kc,   // [B,H,S,D]
    const bf16* __restrict__ Vt,   // [B,H,D,S]
    const float* __restrict__ qvar, const float* __restrict__ kvar,
    bf16* __restrict__ attn_out) { // [B,S,E] bf16
  extern __shared__ float scores[]; // [TQ][SS] = 128 KB of the 320 KB WGP LDS

  const int tid  = threadIdx.x;
  const int lane = tid & 31;
  const int wave = tid >> 5;
  const int qblk = blockIdx.x & (SS / TQ - 1);
  const int bh   = blockIdx.x >> 5;
  const int h    = bh & (HH - 1);
  const int b    = bh >> 4;

  const size_t head = (size_t)bh * SS * DD;
  const int q0 = qblk * TQ;
  const int n16 = lane & 15;
  const int mo  = (lane >= 16) ? 8 : 0;
  const float inv_d = 1.0f / 64.0f;

  // ---- phase 1: scores[q, k] = (Qc.Kc)/d + qvar*kvar/d ----
  const int qi = wave & 1;   // which 16-row q block
  const int kj = wave >> 1;  // which k 16-block (mod 4)
  const bf16* qrow = Qc + head + (size_t)(q0 + qi * 16) * DD;
  bf16x16 aq0 = frag_a(qrow, DD, 0, lane);   // d = 0..31
  bf16x16 aq1 = frag_a(qrow, DD, 32, lane);  // d = 32..63
  float qv[8];
  {
    const float* qvp = qvar + (size_t)bh * SS + q0 + qi * 16 + mo;
#pragma unroll
    for (int r = 0; r < 8; ++r) qv[r] = qvp[r];
  }
  for (int t = 0; t < SS / 64; ++t) {           // 16 chunks of 16 cols
    const int kb = (t * 4 + kj) * 16;
    const bf16* krow = Kc + head + (size_t)kb * DD;
    if (t + 1 < SS / 64) // prefetch next chunk's K rows (global_prefetch)
      __builtin_prefetch(krow + (size_t)64 * DD, 0, 0);
    bf16x16 b0 = frag_b(krow, DD, 0, lane);
    bf16x16 b1 = frag_b(krow, DD, 32, lane);
    f32x8 acc = (f32x8){0,0,0,0,0,0,0,0};
    acc = wmma_bf16(aq0, b0, acc);
    acc = wmma_bf16(aq1, b1, acc);
    const float kv = kvar[(size_t)bh * SS + kb + n16];
#pragma unroll
    for (int r = 0; r < 8; ++r)
      scores[(qi * 16 + mo + r) * SS + kb + n16] = acc[r] * inv_d + qv[r] * kv * inv_d;
  }
  __syncthreads();

  // ---- phase 2: softmax over S for each of the 32 rows (4 rows per wave) ----
#pragma unroll
  for (int rr = 0; rr < 4; ++rr) {
    float* srow = scores + (wave * 4 + rr) * SS;
    float mx = -1e30f;
#pragma unroll
    for (int j = 0; j < SS / 32; ++j) mx = fmaxf(mx, srow[lane + j * 32]);
#pragma unroll
    for (int off = 16; off > 0; off >>= 1) mx = fmaxf(mx, __shfl_xor(mx, off, 32));
    float vals[SS / 32];
    float sum = 0.0f;
#pragma unroll
    for (int j = 0; j < SS / 32; ++j) {
      float e = __expf(srow[lane + j * 32] - mx);
      vals[j] = e; sum += e;
    }
#pragma unroll
    for (int off = 16; off > 0; off >>= 1) sum += __shfl_xor(sum, off, 32);
    const float inv = 1.0f / sum;
#pragma unroll
    for (int j = 0; j < SS / 32; ++j) srow[lane + j * 32] = vals[j] * inv;
  }
  __syncthreads();

  // ---- phase 3: out = probs @ V, wave -> (qi, d-block nj) ----
  const int nj = wave >> 1; // 0..3
  f32x8 oacc = (f32x8){0,0,0,0,0,0,0,0};
  const bf16* vrow = Vt + ((size_t)bh * DD + nj * 16 + n16) * SS; // per-lane V^T row
  const float* prow0 = scores + (qi * 16 + n16) * SS;             // A-frag row (m = lane&15)
  const int akb = (lane < 16) ? 0 : 8;
  const int vkb = (lane < 16) ? 0 : 8;
  for (int k0 = 0; k0 < SS; k0 += 32) {
    bf16x16 a;
    const float* p = prow0 + k0 + akb;
#pragma unroll
    for (int i = 0; i < 8; ++i) { a[i] = (bf16)p[i]; a[i + 8] = (bf16)p[i + 16]; }
    const bf16* vp = vrow + k0 + vkb;
    if (k0 + 32 < SS) __builtin_prefetch(vp + 32, 0, 0); // next V chunk
    bf16x8 lo = *(const bf16x8*)(vp);
    bf16x8 hi = *(const bf16x8*)(vp + 16);
    bf16x16 bv;
#pragma unroll
    for (int i = 0; i < 8; ++i) { bv[i] = lo[i]; bv[i + 8] = hi[i]; }
    oacc = wmma_bf16(a, bv, oacc);
  }
#pragma unroll
  for (int r = 0; r < 8; ++r) {
    size_t o = ((size_t)b * SS + q0 + qi * 16 + mo + r) * EE + (size_t)h * DD + nj * 16 + n16;
    attn_out[o] = (bf16)oacc[r];
  }
}

// ---------------------------- host entry -----------------------------------
extern "C" void kernel_launch(void* const* d_in, const int* in_sizes, int n_in,
                              void* d_out, int out_size, void* d_ws, size_t ws_size,
                              hipStream_t stream) {
  (void)in_sizes; (void)n_in; (void)out_size; (void)ws_size;
  const float* x  = (const float*)d_in[0];
  const float* Wq = (const float*)d_in[1];
  const float* Wk = (const float*)d_in[2];
  const float* Wv = (const float*)d_in[3];
  const float* Wo = (const float*)d_in[4];

  size_t off = 0;
  char* ws = (char*)d_ws;
  auto take = [&](size_t bytes) -> char* {
    char* p = ws + off;
    off += (bytes + 255) & ~(size_t)255;
    return p;
  };
  bf16*  xb   = (bf16*) take((size_t)MTOT * EE * sizeof(bf16));
  bf16*  wqb  = (bf16*) take((size_t)EE * EE * sizeof(bf16));
  bf16*  wkb  = (bf16*) take((size_t)EE * EE * sizeof(bf16));
  bf16*  wvb  = (bf16*) take((size_t)EE * EE * sizeof(bf16));
  bf16*  wob  = (bf16*) take((size_t)EE * EE * sizeof(bf16));
  float* Qf   = (float*)take((size_t)MTOT * EE * sizeof(float));
  float* Kf   = (float*)take((size_t)MTOT * EE * sizeof(float));
  float* Vf   = (float*)take((size_t)MTOT * EE * sizeof(float));
  bf16*  Qc   = (bf16*) take((size_t)BB * HH * SS * DD * sizeof(bf16));
  bf16*  Kc   = (bf16*) take((size_t)BB * HH * SS * DD * sizeof(bf16));
  bf16*  Vt   = (bf16*) take((size_t)BB * HH * SS * DD * sizeof(bf16));
  float* qv   = (float*)take((size_t)BB * HH * SS * sizeof(float));
  float* kv   = (float*)take((size_t)BB * HH * SS * sizeof(float));
  bf16*  atb  = (bf16*) take((size_t)MTOT * EE * sizeof(bf16));

  // 1) fp32 -> bf16
  {
    int n4 = MTOT * EE / 4;
    cvt_f32_to_bf16<<<(n4 + 255) / 256, 256, 0, stream>>>(x, xb, n4);
    int w4 = EE * EE / 4;
    cvt_f32_to_bf16<<<(w4 + 255) / 256, 256, 0, stream>>>(Wq, wqb, w4);
    cvt_f32_to_bf16<<<(w4 + 255) / 256, 256, 0, stream>>>(Wk, wkb, w4);
    cvt_f32_to_bf16<<<(w4 + 255) / 256, 256, 0, stream>>>(Wv, wvb, w4);
    cvt_f32_to_bf16<<<(w4 + 255) / 256, 256, 0, stream>>>(Wo, wob, w4);
  }
  // 2) projections Q/K/V = x @ W^T
  dim3 gg(MTOT / GBM, EE / GBN);
  gemm_bf16_nt<<<gg, 256, 0, stream>>>(xb, wqb, Qf, MTOT, EE, EE);
  gemm_bf16_nt<<<gg, 256, 0, stream>>>(xb, wkb, Kf, MTOT, EE, EE);
  gemm_bf16_nt<<<gg, 256, 0, stream>>>(xb, wvb, Vf, MTOT, EE, EE);
  // 3) center + variance + V transpose
  center_heads<<<BB * HH * SS / 8, 256, 0, stream>>>(Qf, Kf, Vf, Qc, Kc, Vt, qv, kv);
  // 4) attention (scores strip resident in LDS)
  attention_kernel<<<BB * HH * (SS / TQ), 256, (size_t)TQ * SS * sizeof(float), stream>>>(
      Qc, Kc, Vt, qv, kv, atb);
  // 5) output projection -> fp32 d_out
  gemm_bf16_nt<<<gg, 256, 0, stream>>>(atb, wob, (float*)d_out, MTOT, EE, EE);
}
